// MultiheadSelfAttention_7095285973413
// MI455X (gfx1250) — matrix operand order repaired and checked by reading, not compiled, not run on hardware
//
#include <hip/hip_runtime.h>
#include <hip/hip_bf16.h>

// ---------------------------------------------------------------------------
// Multihead self-attention for MI455X (gfx1250, wave32, WMMA bf16 path)
// H=1024, NH=16, HS=64, B=4, S=2048
// ---------------------------------------------------------------------------

#define HDIM   1024
#define NHEADS 16
#define HSIZE  64
#define BATCH  4
#define SEQ    2048
#define MROWS  (BATCH * SEQ)   // 8192

typedef __bf16 bf16x16 __attribute__((ext_vector_type(16)));
typedef float  f32x8   __attribute__((ext_vector_type(8)));

union FragAB {
    bf16x16 v;
    uint4   q[2];
    unsigned int u[8];
};

__device__ __forceinline__ unsigned short f2bf(float f) {
    unsigned int u = __float_as_uint(f);
    u += 0x7FFFu + ((u >> 16) & 1u);      // round-to-nearest-even
    return (unsigned short)(u >> 16);
}

__device__ __forceinline__ f32x8 wmma_bf16(const FragAB& a, const FragAB& b, f32x8 c) {
    return __builtin_amdgcn_wmma_f32_16x16x32_bf16(
        false, a.v, false, b.v, (short)0, c, false, false);
}

// Async DMA: global -> LDS, 16 bytes per lane, tracked by ASYNCcnt.
// GVS mode: 64-bit SGPR base + 32-bit per-lane byte offset.
// LDS address = low 32 bits of flat shared pointer (ISA 10.2 LDS aperture).
__device__ __forceinline__ void async_b128(void* lds_dst, const void* sbase,
                                           unsigned byte_off) {
    unsigned lds = (unsigned)(unsigned long long)lds_dst;
    asm volatile("global_load_async_to_lds_b128 %0, %1, %2"
                 :: "v"(lds), "v"(byte_off), "s"(sbase)
                 : "memory");
}
__device__ __forceinline__ void wait_async0() {
    asm volatile("s_wait_asynccnt 0" ::: "memory");
}

// ---------------------------------------------------------------------------
// fp32 -> bf16 conversion, 4 elements per thread
// ---------------------------------------------------------------------------
__global__ void cvt_f32_bf16(const float* __restrict__ src,
                             unsigned short* __restrict__ dst, int n4) {
    int i = blockIdx.x * blockDim.x + threadIdx.x;
    if (i < n4) {
        float4 v = ((const float4*)src)[i];
        ushort4 o;
        o.x = f2bf(v.x); o.y = f2bf(v.y); o.z = f2bf(v.z); o.w = f2bf(v.w);
        ((ushort4*)dst)[i] = o;
    }
}

// ---------------------------------------------------------------------------
// GEMM: Y[M,N] = A[M,K](bf16) * W[N,K]^T(bf16) + bias[N](f32)
// Block tile 128x128, 8 waves x (32x64), K-step 32.
// Double-buffered async-to-LDS staging: DMA of tile kt+1 overlaps WMMA on kt,
// one barrier per K-step.
// MODE 0: fp32 row-major.  MODE 1: bf16 row-major.  MODE 2: bf16 V-transposed.
// ---------------------------------------------------------------------------
template <int MODE>
__global__ __launch_bounds__(256)
void gemm_xWT_bias(const unsigned short* __restrict__ A,
                   const unsigned short* __restrict__ W,
                   const float* __restrict__ bias,
                   float* __restrict__ outF,
                   unsigned short* __restrict__ outB) {
    constexpr int Kd = HDIM, Nd = HDIM;
    __shared__ unsigned short As[2][128][40];   // 32 cols + pad, 16B-aligned rows
    __shared__ unsigned short Bs[2][128][40];

    const int tid    = threadIdx.x;
    const int lane   = tid & 31;
    const int w      = tid >> 5;
    const int wm     = w >> 1;       // 0..3
    const int wn     = w & 1;        // 0..1
    const int halfid = lane >> 4;
    const int ln     = lane & 15;
    const int mBase  = blockIdx.y * 128;
    const int nBase  = blockIdx.x * 128;

    // staging map: thread covers two 8-bf16 chunks of each 128x32 tile
    const int idx0 = tid * 8;           // chunk c: idx = c*2048 + tid*8
    const int row0 = idx0 >> 5, col0 = idx0 & 31;
    const int idx1 = 2048 + tid * 8;
    const int row1 = idx1 >> 5, col1 = idx1 & 31;

    f32x8 acc[2][4];
    #pragma unroll
    for (int i = 0; i < 2; ++i)
        #pragma unroll
        for (int j = 0; j < 4; ++j)
            #pragma unroll
            for (int e = 0; e < 8; ++e) acc[i][j][e] = 0.f;

    auto issue = [&](int buf, int k0) {
        async_b128(&As[buf][row0][col0], A, (unsigned)(((size_t)(mBase + row0) * Kd + k0 + col0) * 2));
        async_b128(&As[buf][row1][col1], A, (unsigned)(((size_t)(mBase + row1) * Kd + k0 + col1) * 2));
        async_b128(&Bs[buf][row0][col0], W, (unsigned)(((size_t)(nBase + row0) * Kd + k0 + col0) * 2));
        async_b128(&Bs[buf][row1][col1], W, (unsigned)(((size_t)(nBase + row1) * Kd + k0 + col1) * 2));
    };

    const int nkt = Kd / 32;
    issue(0, 0);
    for (int kt = 0; kt < nkt; ++kt) {
        const int cur = kt & 1;
        wait_async0();        // my DMA into buf `cur` complete
        __syncthreads();      // everyone's DMA complete / prev compute done
        if (kt + 1 < nkt) issue(cur ^ 1, (kt + 1) * 32);

        FragAB af[2], wfr[4];
        #pragma unroll
        for (int tm = 0; tm < 2; ++tm) {
            int m = wm * 32 + tm * 16 + ln;
            af[tm].q[0] = *(const uint4*)&As[cur][m][8 * halfid];
            af[tm].q[1] = *(const uint4*)&As[cur][m][16 + 8 * halfid];
        }
        #pragma unroll
        for (int tn = 0; tn < 4; ++tn) {
            int n = wn * 64 + tn * 16 + ln;
            wfr[tn].q[0] = *(const uint4*)&Bs[cur][n][8 * halfid];
            wfr[tn].q[1] = *(const uint4*)&Bs[cur][n][16 + 8 * halfid];
        }
        #pragma unroll
        for (int tm = 0; tm < 2; ++tm)
            #pragma unroll
            for (int tn = 0; tn < 4; ++tn)
                acc[tm][tn] = wmma_bf16(af[tm], wfr[tn], acc[tm][tn]);
    }

    // Epilogue: C layout row = j + 8*half, col = ln
    #pragma unroll
    for (int tm = 0; tm < 2; ++tm) {
        #pragma unroll
        for (int tn = 0; tn < 4; ++tn) {
            int nG = nBase + wn * 64 + tn * 16 + ln;
            float bv = bias[nG];
            int mG0 = mBase + wm * 32 + tm * 16 + 8 * halfid;
            #pragma unroll
            for (int j = 0; j < 8; ++j) {
                int m = mG0 + j;
                float y = acc[tm][tn][j] + bv;
                if (MODE == 0) {
                    outF[(size_t)m * Nd + nG] = y;
                } else if (MODE == 1) {
                    outB[(size_t)m * Nd + nG] = f2bf(y);
                } else {
                    int bb = m >> 11;            // m / SEQ
                    int s  = m & (SEQ - 1);
                    outB[(size_t)bb * HDIM * SEQ + (size_t)nG * SEQ + s] = f2bf(y);
                }
            }
        }
    }
}

// ---------------------------------------------------------------------------
// Flash attention: 4 waves/block, each wave owns 32 queries (2 WMMA q-tiles)
// of one (b,h); streams keys in 64-wide chunks.
//   scores : 16 WMMAs / chunk (Kdim = 64 over 2 frags)
//   softmax: online, row-max via 16-lane shfl butterflies; row-SUM comes free
//            as a 5th ctx accumulator against a constant ones B-matrix.
//   ctx    : 20 WMMAs / chunk against transposed V.
// P tile is wave-private LDS (no block barriers needed).
// ---------------------------------------------------------------------------
__global__ __launch_bounds__(128)
void attention_flash(const unsigned short* __restrict__ Qm,
                     const unsigned short* __restrict__ Km,
                     const unsigned short* __restrict__ Vt,
                     unsigned short* __restrict__ ctx) {
    __shared__ unsigned short Plds[4][32][72];   // per-wave 32q x 64k, padded rows

    const int tid    = threadIdx.x;
    const int lane   = tid & 31;
    const int w      = tid >> 5;
    const int halfid = lane >> 4;
    const int ln     = lane & 15;

    const int qt = blockIdx.x & 15;          // S/128 = 16 query tiles
    const int h  = (blockIdx.x >> 4) & 15;
    const int b  = blockIdx.x >> 8;

    const int qbase = qt * 128 + w * 32;

    const unsigned short* Qp = Qm + (size_t)b * SEQ * HDIM + (size_t)h * HSIZE;
    const unsigned short* Kp = Km + (size_t)b * SEQ * HDIM + (size_t)h * HSIZE;
    const unsigned short* Vp = Vt + (size_t)b * HDIM * SEQ + (size_t)h * HSIZE * SEQ;

    // constant ones B-matrix (bf16 1.0 = 0x3F80) for row-sum accumulation
    FragAB ones;
    #pragma unroll
    for (int i = 0; i < 8; ++i) ones.u[i] = 0x3F803F80u;

    // Q fragments: 2 q-tiles x 2 K-dim frags, loaded once
    FragAB aq[2][2];
    #pragma unroll
    for (int qa = 0; qa < 2; ++qa) {
        const unsigned short* qrow = Qp + (size_t)(qbase + qa * 16 + ln) * HDIM;
        #pragma unroll
        for (int f = 0; f < 2; ++f) {
            aq[qa][f].q[0] = *(const uint4*)(qrow + f * 32 + 8 * halfid);
            aq[qa][f].q[1] = *(const uint4*)(qrow + f * 32 + 16 + 8 * halfid);
        }
    }

    float m_run[2][8];
    f32x8 acc[2][5];                       // [qa][0..3]=ctx dims, [4]=row-sum l
    #pragma unroll
    for (int qa = 0; qa < 2; ++qa) {
        #pragma unroll
        for (int j = 0; j < 8; ++j) m_run[qa][j] = -1e30f;
        #pragma unroll
        for (int t = 0; t < 5; ++t)
            #pragma unroll
            for (int e = 0; e < 8; ++e) acc[qa][t][e] = 0.f;
    }

    const float scale = 0.125f;  // 1/sqrt(HS=64)

    for (int kb = 0; kb < SEQ; kb += 64) {
        // ---- scores: 2 qa x 4 key-subtiles, Kdim = 64
        f32x8 s[2][4];
        #pragma unroll
        for (int kt = 0; kt < 4; ++kt) {
            FragAB bk0, bk1;
            const unsigned short* kr = Kp + (size_t)(kb + kt * 16 + ln) * HDIM + 8 * halfid;
            bk0.q[0] = *(const uint4*)(kr);
            bk0.q[1] = *(const uint4*)(kr + 16);
            bk1.q[0] = *(const uint4*)(kr + 32);
            bk1.q[1] = *(const uint4*)(kr + 48);
            #pragma unroll
            for (int qa = 0; qa < 2; ++qa) {
                f32x8 z;
                #pragma unroll
                for (int e = 0; e < 8; ++e) z[e] = 0.f;
                z = wmma_bf16(aq[qa][0], bk0, z);
                s[qa][kt] = wmma_bf16(aq[qa][1], bk1, z);
            }
        }

        // ---- online softmax (row-max only; row-sum via ones-WMMA)
        #pragma unroll
        for (int qa = 0; qa < 2; ++qa) {
            #pragma unroll
            for (int j = 0; j < 8; ++j) {
                float a0 = s[qa][0][j] * scale, a1 = s[qa][1][j] * scale;
                float a2 = s[qa][2][j] * scale, a3 = s[qa][3][j] * scale;
                float rmax = fmaxf(fmaxf(a0, a1), fmaxf(a2, a3));
                #pragma unroll
                for (int off = 1; off < 16; off <<= 1)
                    rmax = fmaxf(rmax, __shfl_xor(rmax, off, 32));
                float newm  = fmaxf(m_run[qa][j], rmax);
                float alpha = __expf(m_run[qa][j] - newm);
                m_run[qa][j] = newm;
                float p0 = __expf(a0 - newm);
                float p1 = __expf(a1 - newm);
                float p2 = __expf(a2 - newm);
                float p3 = __expf(a3 - newm);
                #pragma unroll
                for (int t = 0; t < 5; ++t) acc[qa][t][j] *= alpha;
                unsigned short* pr = &Plds[w][qa * 16 + j + 8 * halfid][ln];
                pr[0]  = f2bf(p0);
                pr[16] = f2bf(p1);
                pr[32] = f2bf(p2);
                pr[48] = f2bf(p3);
            }
        }

        // ---- ctx += P(32x64) * V(64x64); l += P * ones  (wave-private LDS)
        #pragma unroll
        for (int kf = 0; kf < 2; ++kf) {
            FragAB ap0, ap1;
            ap0.q[0] = *(const uint4*)&Plds[w][ln][kf * 32 + 8 * halfid];
            ap0.q[1] = *(const uint4*)&Plds[w][ln][kf * 32 + 16 + 8 * halfid];
            ap1.q[0] = *(const uint4*)&Plds[w][16 + ln][kf * 32 + 8 * halfid];
            ap1.q[1] = *(const uint4*)&Plds[w][16 + ln][kf * 32 + 16 + 8 * halfid];
            acc[0][4] = wmma_bf16(ap0, ones, acc[0][4]);
            acc[1][4] = wmma_bf16(ap1, ones, acc[1][4]);
            #pragma unroll
            for (int t = 0; t < 4; ++t) {
                FragAB bvf;
                const unsigned short* vr =
                    Vp + (size_t)(t * 16 + ln) * SEQ + kb + kf * 32 + 8 * halfid;
                bvf.q[0] = *(const uint4*)(vr);
                bvf.q[1] = *(const uint4*)(vr + 16);
                acc[0][t] = wmma_bf16(ap0, bvf, acc[0][t]);
                acc[1][t] = wmma_bf16(ap1, bvf, acc[1][t]);
            }
        }
    }

    // ---- normalize by l and store ctx (bf16, [B,S,H] head-interleaved)
    #pragma unroll
    for (int qa = 0; qa < 2; ++qa) {
        #pragma unroll
        for (int t = 0; t < 4; ++t) {
            int d = h * HSIZE + t * 16 + ln;
            #pragma unroll
            for (int j = 0; j < 8; ++j) {
                int q = qbase + qa * 16 + j + 8 * halfid;
                float y = acc[qa][t][j] / acc[qa][4][j];
                ctx[(size_t)(b * SEQ + q) * HDIM + d] = f2bf(y);
            }
        }
    }
}

// ---------------------------------------------------------------------------
// Host-side orchestration
// ---------------------------------------------------------------------------
extern "C" void kernel_launch(void* const* d_in, const int* in_sizes, int n_in,
                              void* d_out, int out_size, void* d_ws, size_t ws_size,
                              hipStream_t stream) {
    const float* x  = (const float*)d_in[0];
    const float* Wq = (const float*)d_in[1];
    const float* bq = (const float*)d_in[2];
    const float* Wk = (const float*)d_in[3];
    const float* bk = (const float*)d_in[4];
    const float* Wv = (const float*)d_in[5];
    const float* bv = (const float*)d_in[6];
    const float* Wo = (const float*)d_in[7];
    const float* bo = (const float*)d_in[8];
    float* out = (float*)d_out;

    char*  ws  = (char*)d_ws;
    size_t off = 0;
    auto alloc = [&](size_t bytes) -> void* {
        void* p = ws + off;
        off += (bytes + 255) & ~(size_t)255;
        return p;
    };
    unsigned short* xb   = (unsigned short*)alloc((size_t)MROWS * HDIM * 2);
    unsigned short* Wqb  = (unsigned short*)alloc((size_t)HDIM * HDIM * 2);
    unsigned short* Wkb  = (unsigned short*)alloc((size_t)HDIM * HDIM * 2);
    unsigned short* Wvb  = (unsigned short*)alloc((size_t)HDIM * HDIM * 2);
    unsigned short* Wob  = (unsigned short*)alloc((size_t)HDIM * HDIM * 2);
    unsigned short* Qb   = (unsigned short*)alloc((size_t)MROWS * HDIM * 2);
    unsigned short* Kb   = (unsigned short*)alloc((size_t)MROWS * HDIM * 2);
    unsigned short* Vtb  = (unsigned short*)alloc((size_t)MROWS * HDIM * 2);
    unsigned short* ctxb = (unsigned short*)alloc((size_t)MROWS * HDIM * 2);

    {
        int n4 = MROWS * HDIM / 4;
        cvt_f32_bf16<<<(n4 + 255) / 256, 256, 0, stream>>>(x, xb, n4);
        n4 = HDIM * HDIM / 4;
        cvt_f32_bf16<<<(n4 + 255) / 256, 256, 0, stream>>>(Wq, Wqb, n4);
        cvt_f32_bf16<<<(n4 + 255) / 256, 256, 0, stream>>>(Wk, Wkb, n4);
        cvt_f32_bf16<<<(n4 + 255) / 256, 256, 0, stream>>>(Wv, Wvb, n4);
        cvt_f32_bf16<<<(n4 + 255) / 256, 256, 0, stream>>>(Wo, Wob, n4);
    }

    dim3 gg(HDIM / 128, MROWS / 128);   // (8, 64)
    gemm_xWT_bias<1><<<gg, 256, 0, stream>>>(xb, Wqb, bq, nullptr, Qb);
    gemm_xWT_bias<1><<<gg, 256, 0, stream>>>(xb, Wkb, bk, nullptr, Kb);
    gemm_xWT_bias<2><<<gg, 256, 0, stream>>>(xb, Wvb, bv, nullptr, Vtb);

    attention_flash<<<BATCH * NHEADS * (SEQ / 128), 128, 0, stream>>>(Qb, Kb, Vtb, ctxb);

    gemm_xWT_bias<0><<<gg, 256, 0, stream>>>(ctxb, Wob, bo, out, nullptr);
}
